// Attention_66340064854151
// MI455X (gfx1250) — compile-verified, matrix operand order checked by reading, not run on hardware
//
#include <hip/hip_runtime.h>
#include <hip/hip_bf16.h>

#define SEQ 4096
#define DMODEL 1024

typedef __bf16 bf16;
typedef __attribute__((ext_vector_type(16))) __bf16 v16bf;
typedef __attribute__((ext_vector_type(8)))  __bf16 v8bf;
typedef __attribute__((ext_vector_type(8)))  float  v8f;

// ---------------------------------------------------------------------------
// f32 -> bf16 conversion, 8 elements per thread (float4 x2 in, v8bf out)
// ---------------------------------------------------------------------------
__global__ __launch_bounds__(256) void cvt_f32_bf16_kernel(
    const float* __restrict__ src, bf16* __restrict__ dst, int n8) {
  int i = blockIdx.x * 256 + threadIdx.x;
  if (i >= n8) return;
  const float4* s = (const float4*)src + (size_t)i * 2;
  float4 a = s[0], b = s[1];
  v8bf o;
  o[0] = (bf16)a.x; o[1] = (bf16)a.y; o[2] = (bf16)a.z; o[3] = (bf16)a.w;
  o[4] = (bf16)b.x; o[5] = (bf16)b.y; o[6] = (bf16)b.z; o[7] = (bf16)b.w;
  *((v8bf*)dst + i) = o;
}

// ---------------------------------------------------------------------------
// CDNA5 async global->LDS copy of one 16-byte chunk per lane.
// GLOBAL_LOAD_ASYNC_TO_LDS_B128 (VGLOBAL op 98): VDST = LDS byte address VGPR,
// VADDR = 64-bit global address. Tracked with ASYNCcnt (in-order completion).
// LDS byte address = low 32 bits of the generic pointer.
// ---------------------------------------------------------------------------
__device__ __forceinline__ void async_ld16(unsigned lds_off,
                                           const void* gaddr) {
  asm volatile("global_load_async_to_lds_b128 %0, %1, off"
               :
               : "v"(lds_off), "v"((unsigned long long)(uintptr_t)gaddr)
               : "memory");
}

// ---------------------------------------------------------------------------
// WMMA fragment load from LDS (row-major, 16 rows x 32 bf16 columns).
// A-layout (ISA 7.12.2, 16-bit A 16x32):
//   lane = h*16 + m ; elems 0..7  <- row m, K = h*8 + 0..7
//                     elems 8..15 <- row m, K = 16 + h*8 + 0..7
// B fragment for NT GEMM (B[k][n] = Bmat[n][k], Bmat row-major) is identical
// with n in place of m.
// ---------------------------------------------------------------------------
__device__ __forceinline__ v16bf lds_frag(const bf16* p, int ld, int lane) {
  int m = lane & 15, h = lane >> 4;
  const bf16* r = p + m * ld + h * 8;
  v8bf lo = *(const v8bf*)(r);
  v8bf hi = *(const v8bf*)(r + 16);
  union { v16bf v; v8bf h2[2]; } u;
  u.h2[0] = lo; u.h2[1] = hi;
  return u.v;
}

// ---------------------------------------------------------------------------
// Generic NT GEMM:  C[M][N] = scale * A[M][K] x B[N][K]^T   (A,B bf16 rowmajor)
// Block tile 128x256, K-step 32, double-buffered LDS filled by async
// global->LDS DMA (ASYNCcnt). Padded LDS rows (40 bf16 = 20-bank stride) give
// conflict-free ds_load_b128 fragment reads. 8 waves: 2(M) x 4(N); each wave
// computes a 64x64 tile = 4x4 fragments -> 16 v_wmma_f32_16x16x32_bf16 per
// K-step with 4x reuse of every fragment and half the barriers per FLOP.
// MODE 0: f32 out rowmajor; MODE 1: bf16 out rowmajor; MODE 2: bf16 out
// transposed (C[n][m], ld = M) -- used to produce V^T directly.
// ---------------------------------------------------------------------------
constexpr int BM = 128, BN = 256, BK = 32, BKP = 40;

template <int MODE>
__global__ __launch_bounds__(256, 1) void gemm_nt_kernel(
    const bf16* __restrict__ A, const bf16* __restrict__ B,
    void* __restrict__ Cout, int M, int N, int K, float scale) {
  __shared__ bf16 sA[2][BM][BKP];
  __shared__ bf16 sB[2][BN][BKP];

  const int tid  = threadIdx.x;
  const int lane = tid & 31;
  const int wave = tid >> 5;
  const int wm   = wave & 1;   // M sub-tile: wm*64
  const int wn   = wave >> 1;  // N sub-tile: wn*64
  const int bm0  = blockIdx.y * BM;
  const int bn0  = blockIdx.x * BN;

  // Copy slots: A = 512 16B-chunks (2/thread), B = 1024 chunks (4/thread).
  const int r0 = tid >> 2;       // 0..63
  const int q  = tid & 3;        // 16B chunk within a 64B row

  const bf16* gA[2];
  const bf16* gB[4];
  unsigned    lA[2][2], lB[2][4];
#pragma unroll
  for (int i = 0; i < 2; ++i) {
    gA[i] = A + (size_t)(bm0 + r0 + i * 64) * K + q * 8;
#pragma unroll
    for (int b = 0; b < 2; ++b)
      lA[b][i] = (unsigned)(uintptr_t)&sA[b][r0 + i * 64][q * 8];
  }
#pragma unroll
  for (int j = 0; j < 4; ++j) {
    gB[j] = B + (size_t)(bn0 + r0 + j * 64) * K + q * 8;
#pragma unroll
    for (int b = 0; b < 2; ++b)
      lB[b][j] = (unsigned)(uintptr_t)&sB[b][r0 + j * 64][q * 8];
  }

  auto fill = [&](int buf, int k0) {
#pragma unroll
    for (int i = 0; i < 2; ++i) async_ld16(lA[buf][i], gA[i] + k0);
#pragma unroll
    for (int j = 0; j < 4; ++j) async_ld16(lB[buf][j], gB[j] + k0);
  };

  v8f acc[4][4] = {};

  fill(0, 0);
  for (int k0 = 0; k0 < K; k0 += BK) {
    const int buf   = (k0 / BK) & 1;
    const bool more = (k0 + BK) < K;
    if (more) {
      fill(buf ^ 1, k0 + BK);                      // DMA next tile
      __builtin_prefetch(gA[0] + k0 + 2 * BK, 0, 1); // warm L2 for K+2
      __builtin_prefetch(gB[0] + k0 + 2 * BK, 0, 1);
      // 12 outstanding; <=6 left means the 6 ops filling `buf` completed.
      asm volatile("s_wait_asynccnt 0x6" ::: "memory");
    } else {
      asm volatile("s_wait_asynccnt 0x0" ::: "memory");
    }
    __syncthreads();  // all waves' fills of `buf` visible

    v16bf afrag[4], bfrag[4];
#pragma unroll
    for (int i = 0; i < 4; ++i)
      afrag[i] = lds_frag(&sA[buf][wm * 64 + i * 16][0], BKP, lane);
#pragma unroll
    for (int j = 0; j < 4; ++j)
      bfrag[j] = lds_frag(&sB[buf][wn * 64 + j * 16][0], BKP, lane);

#pragma unroll
    for (int i = 0; i < 4; ++i)
#pragma unroll
      for (int j = 0; j < 4; ++j)
        acc[i][j] = __builtin_amdgcn_wmma_f32_16x16x32_bf16(
            false, afrag[i], false, bfrag[j], (short)0, acc[i][j], false, false);

    __syncthreads();  // everyone done reading `buf` before it is refilled
  }

  // ---- epilogue: C/D layout: col = lane&15, row = r + 8*(lane>>4) ----
  const int n = lane & 15, h = lane >> 4;
#pragma unroll
  for (int i = 0; i < 4; ++i) {
#pragma unroll
    for (int j = 0; j < 4; ++j) {
      int m0 = bm0 + wm * 64 + i * 16;
      int n0 = bn0 + wn * 64 + j * 16;
#pragma unroll
      for (int r = 0; r < 8; ++r) {
        int row = m0 + r + 8 * h;
        int col = n0 + n;
        float val = acc[i][j][r] * scale;
        if (MODE == 0)
          ((float*)Cout)[(size_t)row * N + col] = val;
        else if (MODE == 1)
          ((bf16*)Cout)[(size_t)row * N + col] = (bf16)val;
        else
          ((bf16*)Cout)[(size_t)col * M + row] = (bf16)val;
      }
    }
  }
}

// ---------------------------------------------------------------------------
// Row softmax: S (f32, already pre-scaled by 1/sqrt(D)) -> P (bf16)
// One 256-thread workgroup per row of 4096.
// ---------------------------------------------------------------------------
__global__ __launch_bounds__(256) void softmax_kernel(
    const float* __restrict__ S, bf16* __restrict__ P, int n) {
  __shared__ float red[256];
  const int row = blockIdx.x;
  const int tid = threadIdx.x;
  const float* x = S + (size_t)row * n;

  float mx = -3.4e38f;
  for (int i = tid; i < n; i += 256) mx = fmaxf(mx, x[i]);
  red[tid] = mx;
  __syncthreads();
  for (int s = 128; s > 0; s >>= 1) {
    if (tid < s) red[tid] = fmaxf(red[tid], red[tid + s]);
    __syncthreads();
  }
  mx = red[0];
  __syncthreads();

  float sum = 0.f;
  for (int i = tid; i < n; i += 256) sum += __expf(x[i] - mx);
  red[tid] = sum;
  __syncthreads();
  for (int s = 128; s > 0; s >>= 1) {
    if (tid < s) red[tid] += red[tid + s];
    __syncthreads();
  }
  float inv = 1.0f / red[0];

  bf16* p = P + (size_t)row * n;
  for (int i = tid; i < n; i += 256) p[i] = (bf16)(__expf(x[i] - mx) * inv);
}

// ---------------------------------------------------------------------------
// Orchestration:
//   bf16-cast inputs -> Q,K (bf16), V^T (bf16) -> S=QK^T/32 (f32, lives in L2)
//   -> softmax -> P (bf16) -> O = P * V  (f32 to d_out)
// ---------------------------------------------------------------------------
extern "C" void kernel_launch(void* const* d_in, const int* in_sizes, int n_in,
                              void* d_out, int out_size, void* d_ws, size_t ws_size,
                              hipStream_t stream) {
  const float* Xq = (const float*)d_in[0];
  const float* Xk = (const float*)d_in[1];
  const float* Xv = (const float*)d_in[2];
  const float* Wq = (const float*)d_in[3];
  const float* Wk = (const float*)d_in[4];
  const float* Wv = (const float*)d_in[5];

  const size_t SD = (size_t)SEQ * DMODEL;   // 4,194,304
  const size_t DD = (size_t)DMODEL * DMODEL;
  const size_t SS = (size_t)SEQ * SEQ;      // 16,777,216

  char* w = (char*)d_ws;
  auto alloc = [&](size_t bytes) -> void* {
    void* p = (void*)w;
    w += (bytes + 255) & ~(size_t)255;
    return p;
  };

  bf16* Xqb = (bf16*)alloc(SD * 2);
  bf16* Xkb = (bf16*)alloc(SD * 2);
  bf16* Xvb = (bf16*)alloc(SD * 2);
  bf16* Wqb = (bf16*)alloc(DD * 2);
  bf16* Wkb = (bf16*)alloc(DD * 2);
  bf16* Wvb = (bf16*)alloc(DD * 2);
  bf16* Qb  = (bf16*)alloc(SD * 2);
  bf16* Kb  = (bf16*)alloc(SD * 2);
  bf16* Vt  = (bf16*)alloc(SD * 2);         // [DMODEL][SEQ]
  bf16* Pb  = (bf16*)alloc(SS * 2);
  float* Sf = (float*)alloc(SS * 4);

  // 1) f32 -> bf16
  cvt_f32_bf16_kernel<<<(int)(SD / 8 / 256), 256, 0, stream>>>(Xq, Xqb, (int)(SD / 8));
  cvt_f32_bf16_kernel<<<(int)(SD / 8 / 256), 256, 0, stream>>>(Xk, Xkb, (int)(SD / 8));
  cvt_f32_bf16_kernel<<<(int)(SD / 8 / 256), 256, 0, stream>>>(Xv, Xvb, (int)(SD / 8));
  cvt_f32_bf16_kernel<<<(int)(DD / 8 / 256), 256, 0, stream>>>(Wq, Wqb, (int)(DD / 8));
  cvt_f32_bf16_kernel<<<(int)(DD / 8 / 256), 256, 0, stream>>>(Wk, Wkb, (int)(DD / 8));
  cvt_f32_bf16_kernel<<<(int)(DD / 8 / 256), 256, 0, stream>>>(Wv, Wvb, (int)(DD / 8));

  // 2) Projections: y = X @ W^T  (torch Linear). NT shape fits directly.
  dim3 gProj(DMODEL / BN, SEQ / BM);  // (4, 32)
  gemm_nt_kernel<1><<<gProj, 256, 0, stream>>>(Xqb, Wqb, Qb, SEQ, DMODEL, DMODEL, 1.0f);
  gemm_nt_kernel<1><<<gProj, 256, 0, stream>>>(Xkb, Wkb, Kb, SEQ, DMODEL, DMODEL, 1.0f);
  gemm_nt_kernel<2><<<gProj, 256, 0, stream>>>(Xvb, Wvb, Vt, SEQ, DMODEL, DMODEL, 1.0f);

  // 3) S = Q K^T / sqrt(D)   (f32, 64 MB -> resident in 192 MB L2)
  dim3 gS(SEQ / BN, SEQ / BM);        // (16, 32)
  gemm_nt_kernel<0><<<gS, 256, 0, stream>>>(Qb, Kb, Sf, SEQ, SEQ, DMODEL, 1.0f / 32.0f);

  // 4) softmax rows -> P bf16
  softmax_kernel<<<SEQ, 256, 0, stream>>>(Sf, Pb, SEQ);

  // 5) O = P @ V = P x (V^T)^T  (NT again, f32 out to d_out)
  dim3 gO(DMODEL / BN, SEQ / BM);     // (4, 32)
  gemm_nt_kernel<0><<<gO, 256, 0, stream>>>(Pb, Vt, (float*)d_out, SEQ, DMODEL, SEQ, 1.0f);
}
